// ExpCModel_7851200217319
// MI455X (gfx1250) — compile-verified, matrix operand order checked by reading, not run on hardware
//
#include <hip/hip_runtime.h>
#include <hip/hip_bf16.h>
#include <math.h>

// ---------------- problem constants ----------------
#define NN   8192
#define DD   768
#define EE   16384
#define LL   3
#define HH   4
#define RR   64
#define HD   (HH*DD)      // 3072
#define NEG  0.2f
#define EPS_LN 1e-5f

// ---------------- WMMA / TDM types ----------------
typedef __attribute__((ext_vector_type(16))) __bf16        v16bf;
typedef __attribute__((ext_vector_type(8)))  float         v8f;
typedef __attribute__((ext_vector_type(4)))  unsigned int  v4u;
typedef __attribute__((ext_vector_type(8)))  int           v8i;
typedef __attribute__((ext_vector_type(4)))  int           v4i;

__device__ __forceinline__ unsigned short f2bf(float f) {
    unsigned u = __builtin_bit_cast(unsigned, f);
    unsigned r = (u + 0x7FFFu + ((u >> 16) & 1u)) >> 16;   // round-to-nearest-even
    return (unsigned short)r;
}
__device__ __forceinline__ unsigned pack2bf(float a, float b) {
    return (unsigned)f2bf(a) | ((unsigned)f2bf(b) << 16);
}
__device__ __forceinline__ __bf16 bfbits(unsigned short s) {
    return __builtin_bit_cast(__bf16, s);
}
__device__ __forceinline__ float gelu_erf(float x) {
    return 0.5f * x * (1.0f + erff(x * 0.70710678118654752f));
}
// The TDM writes LDS behind the compiler's back. Escape the LDS pointer into
// an opaque asm with a memory clobber so the compiler must assume the buffer
// is (re)written and cannot fold the fragment loads to undef.
__device__ __forceinline__ void lds_dma_fence(const void* p) {
    asm volatile("" :: "v"(p) : "memory");
}

// ---------------- GEMM tiling ----------------
#define BM 128
#define BN 128
#define BK 32
#define LDSROW (BK + 8)                    // 40 ushorts = 80B row (64B data + 16B pad)
#define BUFB   (BM * LDSROW * 2)           // 10240 bytes per LDS tile buffer

// ---- TDM: issue a 2D bf16 tile load (rows x BK) global -> LDS ----
// Descriptor bit layout per CDNA5 ISA ch.7/8 (D# groups). data_size=2B,
// pad_enable with 16-dword interval / 4-dword pad reproduces LDSROW stride.
__device__ __forceinline__ void tdm_load_tile(
    unsigned lds_byte_addr, const unsigned short* gptr, int K, int rows, int dimM)
{
    unsigned long long ga = (unsigned long long)(size_t)gptr;
    v4u g0;
    g0[0] = 1u;                                           // count=1 valid user D#
    g0[1] = lds_byte_addr;                                // lds_addr
    g0[2] = (unsigned)(ga & 0xFFFFFFFFu);                 // global_addr[31:0]
    g0[3] = (unsigned)((ga >> 32) & 0x01FFFFFFu)          // global_addr[56:32]
          | (2u << 30);                                   // type=2 (image)
    unsigned td0 = (unsigned)K;                           // tensor_dim0 (elements)
    unsigned td1 = (unsigned)dimM;                        // tensor_dim1 (rows)
    v8i g1;
    g1[0] = (int)((1u << 16)                              // data_size = 2 bytes
                | (1u << 20)                              // pad_enable
                | (3u << 22)                              // pad_interval = 16 dwords
                | (3u << 25));                            // pad_amount   = 4 dwords
    g1[1] = (int)((td0 & 0xFFFFu) << 16);                 // [63:48] tensor_dim0 lo
    g1[2] = (int)((td0 >> 16) | ((td1 & 0xFFFFu) << 16)); // dim0 hi | dim1 lo
    g1[3] = (int)((td1 >> 16) | ((unsigned)BK << 16));    // dim1 hi | tile_dim0=BK
    g1[4] = (int)(unsigned)rows;                          // tile_dim1 ; tile_dim2=0
    g1[5] = (int)(unsigned)K;                             // tensor_dim0_stride lo32
    g1[6] = 0;                                            // stride hi | dim1_stride lo
    g1[7] = 0;
    v4i gz4 = {0, 0, 0, 0};
    v8i gz8 = {0, 0, 0, 0, 0, 0, 0, 0};
    // 6-arg form (clang-23 / therock-10.0 headers)
    __builtin_amdgcn_tensor_load_to_lds(g0, g1, gz4, gz4, gz8, 0);
}

// ---------------- TDM double-buffered bf16 WMMA GEMM ----------------
// C[M,Nt] = A[M,K](bf16) @ Bt[Nt,K](bf16, pre-transposed) + bias, fp32 out.
// M, Nt multiples of 128; K multiple of 32. 8 waves, 32x64 per wave, 8 wmma/K-step.
__global__ __launch_bounds__(256) void gemm_tdm_bf16(
    const unsigned short* __restrict__ A, const unsigned short* __restrict__ Bt,
    const float* __restrict__ bias, float* __restrict__ C,
    int M, int Nt, int K)
{
    __shared__ unsigned short smem[4][BM][LDSROW];  // [0..1]=A bufs, [2..3]=B bufs

    const int bm = blockIdx.y * BM;
    const int bn = blockIdx.x * BN;
    const int tid  = threadIdx.x;
    const int wid  = tid >> 5;
    const int lane = tid & 31;
    const int wm = wid & 3;          // 32-row slice
    const int wn = wid >> 2;         // 64-col slice
    const int l16 = lane & 15;
    const int hlf = lane >> 4;

    // LDS byte offset of smem (low 32 bits of the flat shared address are the
    // LDS offset); the ptrtoint also escapes smem so AA can't prove it unwritten.
    const unsigned lds_base = (unsigned)(unsigned long long)(const void*)&smem[0][0][0];
    if (tid == 0) smem[0][0][0] = 0;   // store-reachable for AA; overwritten by DMA

    const v8f vzero = {};
    v8f acc[2][4];
    #pragma unroll
    for (int i = 0; i < 2; ++i)
        #pragma unroll
        for (int j = 0; j < 4; ++j) acc[i][j] = vzero;

    const unsigned short* Ab = A  + (size_t)bm * K;
    const unsigned short* Bb = Bt + (size_t)bn * K;

    // prologue: DMA first K-tile into buffers 0 (A) / 2 (B)
    if (wid == 0) {
        tdm_load_tile(lds_base + 0u * BUFB, Ab, K, BM, M);
        tdm_load_tile(lds_base + 2u * BUFB, Bb, K, BN, Nt);
        __builtin_amdgcn_s_wait_tensorcnt(0);
    }
    __syncthreads();
    lds_dma_fence(smem);

    int cur = 0;
    for (int k0 = 0; k0 < K; k0 += BK) {
        const bool more = (k0 + BK) < K;
        if (more && wid == 0) {     // overlap next tile's DMA with this tile's WMMAs
            tdm_load_tile(lds_base + (unsigned)(cur ^ 1) * BUFB,       Ab + k0 + BK, K, BM, M);
            tdm_load_tile(lds_base + (unsigned)(2 + (cur ^ 1)) * BUFB, Bb + k0 + BK, K, BN, Nt);
        }

        // fragments per CDNA5 16-bit striping:
        // lanes 0-15: K pairs 0..7 then 16..23 ; lanes 16-31: +8
        v16bf a[2], b[4];
        #pragma unroll
        for (int i = 0; i < 2; ++i) {
            const unsigned short* pr = smem[cur][wm * 32 + i * 16 + l16];
            #pragma unroll
            for (int v = 0; v < 8; ++v) {
                int kk = (v < 4 ? 2 * v : 16 + 2 * (v - 4)) + hlf * 8;
                a[i][2 * v]     = bfbits(pr[kk]);
                a[i][2 * v + 1] = bfbits(pr[kk + 1]);
            }
        }
        #pragma unroll
        for (int j = 0; j < 4; ++j) {
            const unsigned short* pr = smem[2 + cur][wn * 64 + j * 16 + l16];
            #pragma unroll
            for (int v = 0; v < 8; ++v) {
                int kk = (v < 4 ? 2 * v : 16 + 2 * (v - 4)) + hlf * 8;
                b[j][2 * v]     = bfbits(pr[kk]);
                b[j][2 * v + 1] = bfbits(pr[kk + 1]);
            }
        }
        #pragma unroll
        for (int i = 0; i < 2; ++i)
            #pragma unroll
            for (int j = 0; j < 4; ++j)
                acc[i][j] = __builtin_amdgcn_wmma_f32_16x16x32_bf16(
                    false, a[i], false, b[j], (short)0, acc[i][j], false, false);

        if (more) {
            if (wid == 0) __builtin_amdgcn_s_wait_tensorcnt(0);
            __syncthreads();        // DMA visible + all waves done with old buffer
            lds_dma_fence(smem);
            cur ^= 1;
        }
    }

    // C/D layout: VGPR rr, lanes 0-15 -> M=rr, lanes 16-31 -> M=8+rr ; N=lane&15
    #pragma unroll
    for (int i = 0; i < 2; ++i) {
        int r0 = bm + wm * 32 + i * 16 + hlf * 8;
        #pragma unroll
        for (int j = 0; j < 4; ++j) {
            int c = bn + wn * 64 + j * 16 + l16;
            float bv = bias ? bias[c] : 0.f;
            #pragma unroll
            for (int rr = 0; rr < 8; ++rr)
                C[(size_t)(r0 + rr) * Nt + c] = acc[i][j][rr] + bv;
        }
    }
}

// ---------------- fp32-input WMMA GEMM (M-bounded; for the R=64 relation GEMM) ---
template <bool MB>
__global__ __launch_bounds__(256) void gemm_bf16_wmma(
    const float* __restrict__ A, const float* __restrict__ B,
    const float* __restrict__ bias, float* __restrict__ C,
    int M, int Nt, int K)
{
    __shared__ unsigned short As[BM][LDSROW];
    __shared__ unsigned short Bs[BN][LDSROW];   // transposed: Bs[n][k]

    const int bm = blockIdx.y * BM;
    const int bn = blockIdx.x * BN;
    const int tid  = threadIdx.x;
    const int wid  = tid >> 5;
    const int lane = tid & 31;
    const int wm = wid & 3, wn = wid >> 2;
    const int l16 = lane & 15, hlf = lane >> 4;

    const v8f vzero = {};
    v8f acc[2][4];
    #pragma unroll
    for (int i = 0; i < 2; ++i)
        #pragma unroll
        for (int j = 0; j < 4; ++j) acc[i][j] = vzero;

    for (int k0 = 0; k0 < K; k0 += BK) {
        for (int i = tid; i < BM * BK; i += 256) {
            int r = i >> 5, c = i & (BK - 1);
            int gr = bm + r;
            float v = (!MB || gr < M) ? A[(size_t)gr * K + k0 + c] : 0.f;
            As[r][c] = f2bf(v);
        }
        for (int it = 0; it < 4; ++it) {
            int idx = tid + it * 256;
            int row = idx >> 5;
            int c4  = (idx & 31) << 2;
            const float* pb = B + (size_t)(k0 + row) * Nt + bn + c4;
            float4 v = *(const float4*)pb;
            Bs[c4 + 0][row] = f2bf(v.x);
            Bs[c4 + 1][row] = f2bf(v.y);
            Bs[c4 + 2][row] = f2bf(v.z);
            Bs[c4 + 3][row] = f2bf(v.w);
        }
        __syncthreads();

        v16bf a[2], b[4];
        #pragma unroll
        for (int i = 0; i < 2; ++i) {
            const unsigned short* pr = As[wm * 32 + i * 16 + l16];
            #pragma unroll
            for (int v = 0; v < 8; ++v) {
                int kk = (v < 4 ? 2 * v : 16 + 2 * (v - 4)) + hlf * 8;
                a[i][2 * v]     = bfbits(pr[kk]);
                a[i][2 * v + 1] = bfbits(pr[kk + 1]);
            }
        }
        #pragma unroll
        for (int j = 0; j < 4; ++j) {
            const unsigned short* pr = Bs[wn * 64 + j * 16 + l16];
            #pragma unroll
            for (int v = 0; v < 8; ++v) {
                int kk = (v < 4 ? 2 * v : 16 + 2 * (v - 4)) + hlf * 8;
                b[j][2 * v]     = bfbits(pr[kk]);
                b[j][2 * v + 1] = bfbits(pr[kk + 1]);
            }
        }
        #pragma unroll
        for (int i = 0; i < 2; ++i)
            #pragma unroll
            for (int j = 0; j < 4; ++j)
                acc[i][j] = __builtin_amdgcn_wmma_f32_16x16x32_bf16(
                    false, a[i], false, b[j], (short)0, acc[i][j], false, false);
        __syncthreads();
    }

    #pragma unroll
    for (int i = 0; i < 2; ++i) {
        int r0 = bm + wm * 32 + i * 16 + hlf * 8;
        #pragma unroll
        for (int j = 0; j < 4; ++j) {
            int c = bn + wn * 64 + j * 16 + l16;
            float bv = bias ? bias[c] : 0.f;
            #pragma unroll
            for (int rr = 0; rr < 8; ++rr) {
                int r = r0 + rr;
                if (!MB || r < M)
                    C[(size_t)r * Nt + c] = acc[i][j][rr] + bv;
            }
        }
    }
}

// ---------------- fp32 -> bf16 conversions ----------------
__global__ __launch_bounds__(256) void convert_bf16_kernel(
    const float* __restrict__ src, unsigned short* __restrict__ dst, int n4)
{
    int i = blockIdx.x * 256 + threadIdx.x;
    if (i >= n4) return;
    float4 v = ((const float4*)src)[i];
    ((uint2*)dst)[i] = make_uint2(pack2bf(v.x, v.y), pack2bf(v.z, v.w));
}

// src: Rr x Cc fp32 -> dst: Cc x Rr bf16 (both coalesced via 32x32 LDS tile)
__global__ __launch_bounds__(256) void transpose_convert_kernel(
    const float* __restrict__ src, unsigned short* __restrict__ dst,
    int Rr, int Cc)
{
    __shared__ float tile[32][33];
    const int bx = blockIdx.x * 32;   // col base (src)
    const int by = blockIdx.y * 32;   // row base (src)
    const int t = threadIdx.x;
    const int c = t & 31;
    #pragma unroll
    for (int i = 0; i < 4; ++i) {
        int r = (t >> 5) + i * 8;
        tile[r][c] = src[(size_t)(by + r) * Cc + bx + c];
    }
    __syncthreads();
    #pragma unroll
    for (int i = 0; i < 4; ++i) {
        int r = (t >> 5) + i * 8;
        dst[(size_t)(bx + r) * Rr + by + c] = f2bf(tile[c][r]);
    }
}

// ---------------- edge logits: one wave per (edge, head) ----------------
__global__ __launch_bounds__(128) void edge_logits_kernel(
    const float* __restrict__ xl, const float* __restrict__ xr,
    const float* __restrict__ relW,
    const int* __restrict__ src, const int* __restrict__ dst,
    const int* __restrict__ eattr, const float* __restrict__ att,
    float* __restrict__ logits)
{
    const int e = blockIdx.x;
    const int h = threadIdx.x >> 5;
    const int lane = threadIdx.x & 31;
    const int s = src[e], d = dst[e], r = eattr[e];
    const float4* pl = (const float4*)(xl   + (size_t)s * HD + h * DD);
    const float4* pr = (const float4*)(xr   + (size_t)d * HD + h * DD);
    const float4* pe = (const float4*)(relW + (size_t)r * HD + h * DD);
    const float4* pa = (const float4*)(att  + h * DD);
    float sum = 0.f;
    for (int i = lane; i < DD / 4; i += 32) {
        float4 vl = pl[i], vr = pr[i], ve = pe[i], va = pa[i];
        float m0 = vl.x + vr.x + ve.x; m0 = (m0 > 0.f) ? m0 : NEG * m0;
        float m1 = vl.y + vr.y + ve.y; m1 = (m1 > 0.f) ? m1 : NEG * m1;
        float m2 = vl.z + vr.z + ve.z; m2 = (m2 > 0.f) ? m2 : NEG * m2;
        float m3 = vl.w + vr.w + ve.w; m3 = (m3 > 0.f) ? m3 : NEG * m3;
        sum += m0 * va.x + m1 * va.y + m2 * va.z + m3 * va.w;
    }
    #pragma unroll
    for (int off = 16; off; off >>= 1) sum += __shfl_xor(sum, off, 32);
    if (lane == 0) logits[(size_t)e * HH + h] = sum;
}

// ---------------- segment softmax ----------------
__device__ __forceinline__ unsigned orderFloat(float f) {
    unsigned u = __builtin_bit_cast(unsigned, f);
    return (u & 0x80000000u) ? ~u : (u | 0x80000000u);
}
__device__ __forceinline__ float unorderFloat(unsigned u) {
    unsigned v = (u & 0x80000000u) ? (u & 0x7FFFFFFFu) : ~u;
    return __builtin_bit_cast(float, v);
}

__global__ __launch_bounds__(256) void seg_max_kernel(
    const float* __restrict__ logits, const int* __restrict__ dst,
    unsigned* __restrict__ nmax)
{
    int i = blockIdx.x * 256 + threadIdx.x;
    if (i >= EE * HH) return;
    int e = i >> 2, h = i & 3;
    atomicMax(&nmax[(size_t)dst[e] * HH + h], orderFloat(logits[i]));
}

__global__ __launch_bounds__(256) void seg_exp_kernel(
    const float* __restrict__ logits, const int* __restrict__ dst,
    const unsigned* __restrict__ nmax, float* __restrict__ exs,
    float* __restrict__ den)
{
    int i = blockIdx.x * 256 + threadIdx.x;
    if (i >= EE * HH) return;
    int e = i >> 2, h = i & 3;
    float mx = unorderFloat(nmax[(size_t)dst[e] * HH + h]);
    float ex = __expf(logits[i] - mx);
    exs[i] = ex;
    atomicAdd(&den[(size_t)dst[e] * HH + h], ex);
}

// ---------------- aggregation: out[dst] += alpha * xl[src] ----------------
__global__ __launch_bounds__(256) void aggregate_kernel(
    const float* __restrict__ xl, const int* __restrict__ src,
    const int* __restrict__ dst, const float* __restrict__ exs,
    const float* __restrict__ den, float* __restrict__ outbuf)
{
    const int e = blockIdx.x;
    const int s = src[e], d = dst[e];
    __shared__ float salpha[HH];
    if (threadIdx.x < HH)
        salpha[threadIdx.x] = exs[(size_t)e * HH + threadIdx.x] /
                              (den[(size_t)d * HH + threadIdx.x] + 1e-16f);
    __syncthreads();
    const float* px = xl + (size_t)s * HD;
    float* po = outbuf + (size_t)d * HD;
    for (int i = threadIdx.x; i < HD; i += 256) {
        int h = i / DD;
        atomicAdd(&po[i], px[i] * salpha[h]);
    }
}

// ---------------- node update: x_graph = gelu(mean_h out + bout) + x_graph ------
__global__ __launch_bounds__(256) void node_update_kernel(
    const float* __restrict__ outbuf, const float* __restrict__ bout,
    float* __restrict__ x_graph)
{
    int i = blockIdx.x * 256 + threadIdx.x;           // over NN*DD/4 float4s
    if (i >= NN * DD / 4) return;
    int n  = i / (DD / 4);
    int d4 = i - n * (DD / 4);
    const float4* po = (const float4*)(outbuf + (size_t)n * HD) + d4;
    float4 acc = {0.f, 0.f, 0.f, 0.f};
    #pragma unroll
    for (int h = 0; h < HH; ++h) {
        float4 t = po[h * (DD / 4)];
        acc.x += t.x; acc.y += t.y; acc.z += t.z; acc.w += t.w;
    }
    float4 bo = ((const float4*)bout)[d4];
    float4* pg = (float4*)x_graph + i;
    float4 g = *pg;
    g.x += gelu_erf(acc.x * (1.0f / HH) + bo.x);
    g.y += gelu_erf(acc.y * (1.0f / HH) + bo.y);
    g.z += gelu_erf(acc.z * (1.0f / HH) + bo.z);
    g.w += gelu_erf(acc.w * (1.0f / HH) + bo.w);
    *pg = g;
}

// ---------------- gate + fuse: one wave per node ----------------
__global__ __launch_bounds__(256) void gate_fuse_kernel(
    const float* __restrict__ x_text, const float* __restrict__ x_graph,
    const float* __restrict__ Wg, const float* __restrict__ bg,
    float* __restrict__ x_fused)
{
    const int node = blockIdx.x * 8 + (threadIdx.x >> 5);
    const int lane = threadIdx.x & 31;
    if (node >= NN) return;
    const float* pt = x_text  + (size_t)node * DD;
    const float* pg = x_graph + (size_t)node * DD;
    float sum = 0.f;
    for (int i = lane; i < DD; i += 32)
        sum += pt[i] * Wg[i] + pg[i] * Wg[DD + i];
    #pragma unroll
    for (int off = 16; off; off >>= 1) sum += __shfl_xor(sum, off, 32);
    float alpha = 1.0f / (1.0f + __expf(-(sum + bg[0])));
    float* pf = x_fused + (size_t)node * DD;
    for (int i = lane; i < DD; i += 32)
        pf[i] = alpha * pg[i] + (1.0f - alpha) * pt[i];
}

// ---------------- LayerNorm + GELU: one block per row ----------------
__global__ __launch_bounds__(256) void ln_gelu_kernel(
    const float* __restrict__ y, const float* __restrict__ gamma,
    const float* __restrict__ beta, float* __restrict__ out)
{
    const int row = blockIdx.x;
    const float* py = y + (size_t)row * DD;
    __shared__ float red[8];
    float s = 0.f, ss = 0.f;
    for (int i = threadIdx.x; i < DD; i += 256) {
        float v = py[i];
        s += v; ss += v * v;
    }
    #pragma unroll
    for (int off = 16; off; off >>= 1) {
        s  += __shfl_xor(s,  off, 32);
        ss += __shfl_xor(ss, off, 32);
    }
    if ((threadIdx.x & 31) == 0) red[threadIdx.x >> 5] = s;
    __syncthreads();
    if (threadIdx.x < 8) s = red[threadIdx.x];
    __syncthreads();
    if ((threadIdx.x & 31) == 0) red[threadIdx.x >> 5] = ss;
    __syncthreads();
    if (threadIdx.x < 8) ss = red[threadIdx.x];
    if (threadIdx.x < 8) {
        #pragma unroll
        for (int off = 4; off; off >>= 1) {
            s  += __shfl_xor(s,  off, 32);
            ss += __shfl_xor(ss, off, 32);
        }
        red[0] = s; red[1] = ss;
    }
    __syncthreads();
    float mu  = red[0] * (1.0f / DD);
    float var = red[1] * (1.0f / DD) - mu * mu;
    float inv = rsqrtf(var + EPS_LN);
    float* po = out + (size_t)row * DD;
    for (int i = threadIdx.x; i < DD; i += 256) {
        float v = (py[i] - mu) * inv * gamma[i] + beta[i];
        po[i] = gelu_erf(v);
    }
}

// ---------------- host-side orchestration ----------------
extern "C" void kernel_launch(void* const* d_in, const int* in_sizes, int n_in,
                              void* d_out, int out_size, void* d_ws, size_t ws_size,
                              hipStream_t stream) {
    const float* x_text  = (const float*)d_in[0];
    const float* rel_emb = (const float*)d_in[1];
    const float* Wl      = (const float*)d_in[2];
    const float* bl      = (const float*)d_in[3];
    const float* Wr      = (const float*)d_in[4];
    const float* br      = (const float*)d_in[5];
    const float* We      = (const float*)d_in[6];
    const float* att     = (const float*)d_in[7];
    const float* bout    = (const float*)d_in[8];
    const float* Wg      = (const float*)d_in[9];
    const float* bg      = (const float*)d_in[10];
    const float* Wp      = (const float*)d_in[11];
    const float* bp      = (const float*)d_in[12];
    const float* gamma   = (const float*)d_in[13];
    const float* beta    = (const float*)d_in[14];
    const int*   eidx    = (const int*)d_in[15];
    const int*   eattr   = (const int*)d_in[16];
    const int* src = eidx;
    const int* dst = eidx + EE;

    // workspace layout
    char* ws = (char*)d_ws;
    size_t off = 0;
    auto alloc = [&](size_t bytes) { char* p = ws + off; off += (bytes + 255) & ~size_t(255); return p; };
    float*    xl      = (float*)alloc((size_t)NN * HD * 4);
    float*    xr      = (float*)alloc((size_t)NN * HD * 4);
    float*    outbuf  = (float*)alloc((size_t)NN * HD * 4);
    float*    x_graph = (float*)alloc((size_t)NN * DD * 4);
    float*    relW    = (float*)alloc((size_t)RR * HD * 4);
    float*    logits  = (float*)alloc((size_t)EE * HH * 4);
    float*    exs     = (float*)alloc((size_t)EE * HH * 4);
    unsigned* nmax    = (unsigned*)alloc((size_t)NN * HH * 4);
    float*    den     = (float*)alloc((size_t)NN * HH * 4);
    float*    x_fused = (float*)alloc((size_t)NN * DD * 4);
    float*    ybuf    = (float*)alloc((size_t)NN * DD * 4);
    unsigned short* xg_bf = (unsigned short*)alloc((size_t)NN * DD * 2);  // bf16 x_graph
    unsigned short* xf_bf = (unsigned short*)alloc((size_t)NN * DD * 2);  // bf16 x_fused
    unsigned short* WlT   = (unsigned short*)alloc((size_t)DD * HD * 2);  // bf16 Wl^T [HD][DD]
    unsigned short* WrT   = (unsigned short*)alloc((size_t)DD * HD * 2);  // bf16 Wr^T
    unsigned short* WpT   = (unsigned short*)alloc((size_t)DD * DD * 2);  // bf16 Wp^T

    (void)hipMemcpyAsync(x_graph, x_text, (size_t)NN * DD * 4, hipMemcpyDeviceToDevice, stream);

    dim3 gNode(HD / BN, NN / BM);   // 24 x 64
    dim3 gRel (HD / BN, 1);         // M=64 bounds path
    dim3 gProj(DD / BN, NN / BM);   // 6 x 64
    dim3 gTrW (HD / 32, DD / 32);   // transpose [DD][HD]
    dim3 gTrP (DD / 32, DD / 32);   // transpose [DD][DD]
    const int x4 = NN * DD / 4;

    for (int l = 0; l < LL; ++l) {
        const float* Wl_l = Wl + (size_t)l * DD * HD;
        const float* Wr_l = Wr + (size_t)l * DD * HD;
        const float* We_l = We + (size_t)l * DD * HD;

        convert_bf16_kernel<<<(x4 + 255) / 256, 256, 0, stream>>>(x_graph, xg_bf, x4);
        transpose_convert_kernel<<<gTrW, 256, 0, stream>>>(Wl_l, WlT, DD, HD);
        transpose_convert_kernel<<<gTrW, 256, 0, stream>>>(Wr_l, WrT, DD, HD);

        gemm_tdm_bf16<<<gNode, 256, 0, stream>>>(xg_bf, WlT, bl + (size_t)l * HD,
                                                 xl, NN, HD, DD);
        gemm_tdm_bf16<<<gNode, 256, 0, stream>>>(xg_bf, WrT, br + (size_t)l * HD,
                                                 xr, NN, HD, DD);
        gemm_bf16_wmma<true><<<gRel, 256, 0, stream>>>(rel_emb, We_l, nullptr,
                                                       relW, RR, HD, DD);

        edge_logits_kernel<<<EE, 128, 0, stream>>>(xl, xr, relW, src, dst, eattr,
                                                   att + (size_t)l * HH * DD, logits);

        (void)hipMemsetAsync(nmax,   0, (size_t)NN * HH * 4, stream);
        (void)hipMemsetAsync(den,    0, (size_t)NN * HH * 4, stream);
        (void)hipMemsetAsync(outbuf, 0, (size_t)NN * HD * 4, stream);

        seg_max_kernel<<<(EE * HH + 255) / 256, 256, 0, stream>>>(logits, dst, nmax);
        seg_exp_kernel<<<(EE * HH + 255) / 256, 256, 0, stream>>>(logits, dst, nmax, exs, den);
        aggregate_kernel<<<EE, 256, 0, stream>>>(xl, src, dst, exs, den, outbuf);
        node_update_kernel<<<(NN * DD / 4 + 255) / 256, 256, 0, stream>>>(
            outbuf, bout + (size_t)l * DD, x_graph);
    }

    gate_fuse_kernel<<<NN / 8, 256, 0, stream>>>(x_text, x_graph, Wg, bg, x_fused);
    convert_bf16_kernel<<<(x4 + 255) / 256, 256, 0, stream>>>(x_fused, xf_bf, x4);
    transpose_convert_kernel<<<gTrP, 256, 0, stream>>>(Wp, WpT, DD, DD);
    gemm_tdm_bf16<<<gProj, 256, 0, stream>>>(xf_bf, WpT, bp, ybuf, NN, DD, DD);
    ln_gelu_kernel<<<NN, 256, 0, stream>>>(ybuf, gamma, beta, (float*)d_out);
}
